// LinearMultiHeadAttention_44083544326274
// MI455X (gfx1250) — matrix operand order repaired and checked by reading, not compiled, not run on hardware
//
#include <hip/hip_runtime.h>

// ---------------------------------------------------------------------------
// LinearMultiHeadAttention for MI455X (gfx1250, wave32, WMMA + async/TDM)
//   out = ((x Wq^T) @ ( (x Wk^T)^T (x Wv^T) * scale )) Wo^T      (per b,h)
// All matmuls on v_wmma_f32_16x16x32_bf16; intermediates bf16 (L2-resident).
// GEMM tiles staged with GLOBAL_LOAD_ASYNC_TO_LDS (double buffered); attn's
// q tile staged by the Tensor Data Mover when the builtins are available.
// ---------------------------------------------------------------------------

typedef __attribute__((ext_vector_type(16))) __bf16 v16bf;
typedef __attribute__((ext_vector_type(8)))  __bf16 v8bf;
typedef __attribute__((ext_vector_type(4)))  __bf16 v4bf;
typedef __attribute__((ext_vector_type(8)))  float  v8f;
typedef __attribute__((ext_vector_type(4)))  unsigned int u32x4;
typedef __attribute__((ext_vector_type(8)))  int i32x8;
typedef __attribute__((ext_vector_type(4)))  int i32x4;
typedef int gv4i __attribute__((vector_size(16)));   // matches builtin proto

#define AS1 __attribute__((address_space(1)))
#define AS3 __attribute__((address_space(3)))

#if defined(__HIP_DEVICE_COMPILE__) && \
    __has_builtin(__builtin_amdgcn_global_load_async_to_lds_b128) && \
    __has_builtin(__builtin_amdgcn_s_wait_asynccnt)
#define USE_ASYNC_LDS 1
#else
#define USE_ASYNC_LDS 0
#endif

#if defined(__HIP_DEVICE_COMPILE__) && \
    __has_builtin(__builtin_amdgcn_tensor_load_to_lds) && \
    __has_builtin(__builtin_amdgcn_s_wait_tensorcnt)
#define USE_TDM 1
#else
#define USE_TDM 0
#endif

namespace {
constexpr int kDModel = 1024;
constexpr int kHeads  = 16;
constexpr int kDk     = 64;
constexpr int kBatch  = 4;
constexpr int kSeq    = 4096;
constexpr int kM      = kBatch * kSeq;    // 16384 rows
constexpr int kBH     = kBatch * kHeads;  // 64
constexpr int kSplit  = 16;               // S-splits for the K^T V reduction
}

static __device__ __forceinline__ v16bf cat8(v8bf a, v8bf b) {
  return __builtin_shufflevector(a, b, 0,1,2,3,4,5,6,7,8,9,10,11,12,13,14,15);
}

#if USE_ASYNC_LDS
// 32 contiguous bytes: global -> LDS, no VGPR round-trip (ASYNCcnt tracked).
static __device__ __forceinline__ void async_copy32(const __bf16* g, __bf16* l) {
  __builtin_amdgcn_global_load_async_to_lds_b128(
      (AS1 gv4i*)(void*)g, (AS3 gv4i*)(void*)l, 0, 0);
  __builtin_amdgcn_global_load_async_to_lds_b128(
      (AS1 gv4i*)(void*)g, (AS3 gv4i*)(void*)l, 16, 0);
}
#endif

// ---------------------------- fp32 -> bf16 ---------------------------------
__global__ void f32_to_bf16_kernel(const float4* __restrict__ in,
                                   __bf16* __restrict__ out, int n4) {
  int i = blockIdx.x * blockDim.x + threadIdx.x;
  int stride = gridDim.x * blockDim.x;
  for (; i < n4; i += stride) {
    float4 f = in[i];
    v4bf o = { (__bf16)f.x, (__bf16)f.y, (__bf16)f.z, (__bf16)f.w };
    *(v4bf*)(out + 4 * (size_t)i) = o;
  }
}

// --------------------- big GEMM:  C[M,N] = A[M,K] * W[N,K]^T ----------------
// A row-major bf16, W row-major bf16 (nn.Linear layout). 128x128x32 tiles,
// 256 threads = 8 waves; wave (wm,wn) owns a 32x64 patch = 2x4 wmma tiles.
// Async path: double-buffered LDS, tile k+1 in flight while tile k computes.
template <bool OUT_BF16>
__global__ __launch_bounds__(256)
void gemm_wmma_bf16(const __bf16* __restrict__ A, const __bf16* __restrict__ W,
                    void* __restrict__ Cout, int M, int N, int K) {
  constexpr int BM = 128, BN = 128, BK = 32;
#if USE_ASYNC_LDS
  __shared__ __bf16 As[2][BM][BK];   // 16 KB
  __shared__ __bf16 Ws[2][BN][BK];   // 16 KB
#else
  __shared__ __bf16 As[1][BM][BK];
  __shared__ __bf16 Ws[1][BN][BK];
#endif

  const int t    = threadIdx.x;
  const int lane = t & 31;
  const int wave = t >> 5;
  const int wm   = wave & 3;   // 4 wave-rows (M)
  const int wn   = wave >> 2;  // 2 wave-cols (N)
  const int lh   = lane & 15;
  const int hi   = lane >> 4;  // 0: K-low half, 1: K-high half (contiguous k order)

  const int m0 = blockIdx.y * BM;
  const int n0 = blockIdx.x * BN;

  const int lr = t >> 1;          // 0..127 tile row loaded by this thread
  const int lc = (t & 1) * 16;    // 0 or 16 (halves)

  const __bf16* gA = A + (size_t)(m0 + lr) * K + lc;
  const __bf16* gW = W + (size_t)(n0 + lr) * K + lc;

  const v8f vzero = {0.f,0.f,0.f,0.f,0.f,0.f,0.f,0.f};
  v8f acc[2][4];
#pragma unroll
  for (int i = 0; i < 2; ++i)
#pragma unroll
    for (int j = 0; j < 4; ++j) acc[i][j] = vzero;

  const int nk = K / BK;

#if USE_ASYNC_LDS
  async_copy32(gA, &As[0][lr][lc]);
  async_copy32(gW, &Ws[0][lr][lc]);
#endif

  for (int kt = 0; kt < nk; ++kt) {
    int cb = 0;
#if USE_ASYNC_LDS
    cb = kt & 1;
    __builtin_amdgcn_s_wait_asynccnt(0);   // tile kt fully in LDS
    __syncthreads();                       // ...for every wave in the block
    if (kt + 1 < nk) {                     // overlap: issue tile kt+1 now
      const int nb = (kt + 1) & 1;
      async_copy32(gA + (kt + 1) * BK, &As[nb][lr][lc]);
      async_copy32(gW + (kt + 1) * BK, &Ws[nb][lr][lc]);
    }
#else
    {
      const int k0 = kt * BK;
      const uint4* ga = (const uint4*)(gA + k0);
      *(uint4*)&As[0][lr][lc]     = ga[0];
      *(uint4*)&As[0][lr][lc + 8] = ga[1];
      const uint4* gw = (const uint4*)(gW + k0);
      *(uint4*)&Ws[0][lr][lc]     = gw[0];
      *(uint4*)&Ws[0][lr][lc + 8] = gw[1];
      if (k0 + BK < K) {
        __builtin_prefetch(gA + k0 + BK, 0, 0);
        __builtin_prefetch(gW + k0 + BK, 0, 0);
      }
      __syncthreads();
    }
#endif

    v16bf afrag[2], bfrag[4];
#pragma unroll
    for (int tm = 0; tm < 2; ++tm) {
      const __bf16* pa = &As[cb][wm * 32 + tm * 16 + lh][hi * 16];
      afrag[tm] = *(const v16bf*)pa;
    }
#pragma unroll
    for (int tn = 0; tn < 4; ++tn)
      bfrag[tn] = *(const v16bf*)&Ws[cb][wn * 64 + tn * 16 + lh][hi * 16];

#pragma unroll
    for (int tm = 0; tm < 2; ++tm)
#pragma unroll
      for (int tn = 0; tn < 4; ++tn)
        acc[tm][tn] = __builtin_amdgcn_wmma_f32_16x16x32_bf16(
            false, afrag[tm], false, bfrag[tn], (short)0, acc[tm][tn],
            false, false);
#if !USE_ASYNC_LDS
    __syncthreads();
#endif
  }

#pragma unroll
  for (int tm = 0; tm < 2; ++tm)
#pragma unroll
    for (int tn = 0; tn < 4; ++tn) {
      const int gn = n0 + wn * 64 + tn * 16 + lh;
      const int gm = m0 + wm * 32 + tm * 16 + hi * 8;
      if constexpr (OUT_BF16) {
        __bf16* C = (__bf16*)Cout;
#pragma unroll
        for (int i = 0; i < 8; ++i)
          C[(size_t)(gm + i) * N + gn] = (__bf16)acc[tm][tn][i];
      } else {
        float* C = (float*)Cout;
#pragma unroll
        for (int i = 0; i < 8; ++i)
          C[(size_t)(gm + i) * N + gn] = acc[tm][tn][i];
      }
    }
}

// ------------------- kv partial:  kv[d,e] += sum_s k[s,d] v[s,e] ------------
// One block per (b*h, s-split); 128 threads = 4 waves; wave w owns d-rows
// [16w,16w+16) x all 64 e. K/V chunks staged transposed in LDS so both the
// A (=k^T) and B (=v) fragments are contiguous 16-k runs (k = 16*hi + s order).
__global__ __launch_bounds__(128)
void kv_partial_kernel(const __bf16* __restrict__ kmat,
                       const __bf16* __restrict__ vmat,
                       float* __restrict__ kvpart) {
  __shared__ __bf16 KsT[kDk][32];  // [d][s_local]
  __shared__ __bf16 VsT[kDk][32];  // [e][s_local]

  const int bh = blockIdx.x;      // 0..63
  const int sp = blockIdx.y;      // 0..kSplit-1
  const int b  = bh >> 4, h = bh & 15;
  const int t    = threadIdx.x;
  const int lane = t & 31;
  const int wave = t >> 5;        // tile row (d group)
  const int lh   = lane & 15;
  const int hi   = lane >> 4;

  const int sl = t >> 2;          // 0..31: s row this thread loads
  const int d0 = (t & 3) * 16;    // 16 channels per thread

  const int sBase = sp * (kSeq / kSplit);  // 256 s per split

  const v8f vzero = {0.f,0.f,0.f,0.f,0.f,0.f,0.f,0.f};
  v8f acc[4] = {vzero, vzero, vzero, vzero};

  for (int kt = 0; kt < (kSeq / kSplit) / 32; ++kt) {  // 8 steps of 32 s
    const int s = sBase + kt * 32 + sl;
    const size_t base = (((size_t)b * kSeq + s) * kHeads + h) * kDk + d0;
    v8bf k0 = *(const v8bf*)(kmat + base);
    v8bf k1 = *(const v8bf*)(kmat + base + 8);
    v8bf w0 = *(const v8bf*)(vmat + base);
    v8bf w1 = *(const v8bf*)(vmat + base + 8);

    __syncthreads();  // previous iteration's LDS reads done
#pragma unroll
    for (int j = 0; j < 8; ++j) {
      KsT[d0 + j][sl]     = k0[j];
      KsT[d0 + 8 + j][sl] = k1[j];
      VsT[d0 + j][sl]     = w0[j];
      VsT[d0 + 8 + j][sl] = w1[j];
    }
    __syncthreads();

    const __bf16* pa = &KsT[wave * 16 + lh][hi * 16];
    v16bf afrag = *(const v16bf*)pa;
#pragma unroll
    for (int tn = 0; tn < 4; ++tn) {
      v16bf bfrag = *(const v16bf*)&VsT[tn * 16 + lh][hi * 16];
      acc[tn] = __builtin_amdgcn_wmma_f32_16x16x32_bf16(
          false, afrag, false, bfrag, (short)0, acc[tn], false, false);
    }
  }

  float* outp = kvpart + ((size_t)bh * kSplit + sp) * kDk * kDk;
#pragma unroll
  for (int tn = 0; tn < 4; ++tn) {
    const int gn = tn * 16 + lh;
    const int gm = wave * 16 + hi * 8;
#pragma unroll
    for (int i = 0; i < 8; ++i)
      outp[(gm + i) * kDk + gn] = acc[tn][i];
  }
}

// --------- reduce splits, apply scale=1/sqrt(64), store kv TRANSPOSED ------
__global__ void kv_reduce_kernel(const float* __restrict__ kvpart,
                                 __bf16* __restrict__ kvT) {
  const int bh = blockIdx.x;
  for (int idx = threadIdx.x; idx < kDk * kDk; idx += blockDim.x) {
    const int d = idx >> 6, e = idx & 63;
    const float* p = kvpart + (size_t)bh * kSplit * kDk * kDk + idx;
    float s = 0.f;
#pragma unroll
    for (int j = 0; j < kSplit; ++j) s += p[j * kDk * kDk];
    kvT[(size_t)bh * kDk * kDk + e * kDk + d] = (__bf16)(s * 0.125f);
  }
}

// ---------------- attn[s,e] = sum_d q[s,d] * kv[d,e]   (per b,h) ------------
// 256 threads = 8 waves; wave w owns 16 s-rows x 64 e. kv is stored
// transposed so B-fragments are contiguous 32B global loads (L2 resident).
// The 128x64 q tile is a perfect 2D TDM tile: one wave programs a D# and the
// Tensor Data Mover DMAs it into LDS (TENSORcnt), no VGPR staging.
__global__ __launch_bounds__(256)
void attn_wmma_kernel(const __bf16* __restrict__ q,
                      const __bf16* __restrict__ kvT,
                      __bf16* __restrict__ attn) {
  __shared__ __bf16 Qs[128][kDk];  // 16 KB

  const int bh = blockIdx.y;
  const int b  = bh >> 4, h = bh & 15;
  const int s0 = blockIdx.x * 128;
  const int t    = threadIdx.x;
  const int lane = t & 31;
  const int wave = t >> 5;
  const int lh   = lane & 15;
  const int hi   = lane >> 4;

#if USE_TDM
  if (wave == 0) {
    const unsigned long long gaddr =
        (unsigned long long)(const void*)(q + (((size_t)b * kSeq + s0) * kHeads + h) * kDk);
    const unsigned ldsOff = (unsigned)(size_t)(AS3 __bf16*)(void*)&Qs[0][0];
    // D# group0: count=1, lds_addr, 57b global addr, type=2 (ISA 8.3)
    u32x4 g0;
    g0[0] = 1u;
    g0[1] = ldsOff;
    g0[2] = (unsigned)(gaddr & 0xffffffffu);
    g0[3] = (unsigned)((gaddr >> 32) & 0x01ffffffu) | (2u << 30);
    // D# group1: data_size=2B, tensor 64x128 (dim0 x dim1), tile 64x128,
    // dim0 stride = 1024 halves (H*Dk)                        (ISA 8.4)
    i32x8 g1;
    g1[0] = (int)(1u << 16);          // workgroup_mask=0 | data_size=1 (2B)
    g1[1] = (int)(64u << 16);         // tensor_dim0[15:0]  @ bits 63:48
    g1[2] = (int)(128u << 16);        // tensor_dim1[15:0]  @ bits 95:80
    g1[3] = (int)(64u << 16);         // tile_dim0          @ bits 127:112
    g1[4] = 128;                      // tile_dim1          @ bits 143:128
    g1[5] = 1024;                     // tensor_dim0_stride @ bits 191:160
    g1[6] = 0;
    g1[7] = 0;
    i32x4 z4 = {0, 0, 0, 0};
#if __clang_major__ >= 23
    i32x8 z8 = {0, 0, 0, 0, 0, 0, 0, 0};
    __builtin_amdgcn_tensor_load_to_lds(g0, g1, z4, z4, z8, 0);
#else
    __builtin_amdgcn_tensor_load_to_lds(g0, g1, z4, z4, 0);
#endif
    __builtin_amdgcn_s_wait_tensorcnt(0);
  }
  __syncthreads();
#else
  {  // stage the 128x64 q tile: 64B per thread
    const int r  = t >> 1;
    const int c0 = (t & 1) * 32;
    const uint4* g = (const uint4*)(q + (((size_t)b * kSeq + s0 + r) * kHeads + h) * kDk + c0);
    *(uint4*)&Qs[r][c0]      = g[0];
    *(uint4*)&Qs[r][c0 + 8]  = g[1];
    *(uint4*)&Qs[r][c0 + 16] = g[2];
    *(uint4*)&Qs[r][c0 + 24] = g[3];
  }
  __syncthreads();
#endif

  const v8f vzero = {0.f,0.f,0.f,0.f,0.f,0.f,0.f,0.f};
  v8f acc[4] = {vzero, vzero, vzero, vzero};

#pragma unroll
  for (int kk = 0; kk < 2; ++kk) {  // K = 64 = 2 x 32
    const __bf16* pa = &Qs[wave * 16 + lh][kk * 32 + hi * 16];
    v16bf afrag = *(const v16bf*)pa;
#pragma unroll
    for (int tn = 0; tn < 4; ++tn) {
      const __bf16* pb = kvT + (size_t)bh * kDk * kDk + (tn * 16 + lh) * kDk + kk * 32 + hi * 16;
      v16bf bfrag = *(const v16bf*)pb;
      acc[tn] = __builtin_amdgcn_wmma_f32_16x16x32_bf16(
          false, afrag, false, bfrag, (short)0, acc[tn], false, false);
    }
  }

#pragma unroll
  for (int tn = 0; tn < 4; ++tn) {
    const int gn = h * kDk + tn * 16 + lh;         // column in [B,S,H*Dk]
    const int gm = s0 + wave * 16 + hi * 8;        // s row
#pragma unroll
    for (int i = 0; i < 8; ++i)
      attn[((size_t)b * kSeq + gm + i) * kDModel + gn] = (__bf16)acc[tn][i];
  }
}

// ---------------------------------------------------------------------------
extern "C" void kernel_launch(void* const* d_in, const int* in_sizes, int n_in,
                              void* d_out, int out_size, void* d_ws, size_t ws_size,
                              hipStream_t stream) {
  const float* x  = (const float*)d_in[0];
  const float* Wq = (const float*)d_in[1];
  const float* Wk = (const float*)d_in[2];
  const float* Wv = (const float*)d_in[3];
  const float* Wo = (const float*)d_in[4];
  float* out = (float*)d_out;
  (void)in_sizes; (void)n_in; (void)out_size; (void)ws_size;

  char* ws = (char*)d_ws;
  size_t off = 0;
  auto carve = [&](size_t bytes) {
    char* p = ws + off;
    off += (bytes + 255) & ~(size_t)255;
    return p;
  };
  const size_t actBytes = (size_t)kM * kDModel * sizeof(__bf16);       // 32 MB
  const size_t wBytes   = (size_t)kDModel * kDModel * sizeof(__bf16);  // 2 MB

  __bf16* xb  = (__bf16*)carve(actBytes);
  __bf16* wqb = (__bf16*)carve(wBytes);
  __bf16* wkb = (__bf16*)carve(wBytes);
  __bf16* wvb = (__bf16*)carve(wBytes);
  __bf16* wob = (__bf16*)carve(wBytes);
  __bf16* qb  = (__bf16*)carve(actBytes);
  __bf16* kb  = (__bf16*)carve(actBytes);
  __bf16* vb  = (__bf16*)carve(actBytes);
  float*  kvpart = (float*)carve((size_t)kBH * kSplit * kDk * kDk * sizeof(float));
  __bf16* kvT    = (__bf16*)carve((size_t)kBH * kDk * kDk * sizeof(__bf16));
  __bf16* attnb  = xb;  // x no longer needed once q/k/v exist

  // 1) downconvert inputs to bf16
  f32_to_bf16_kernel<<<1024, 256, 0, stream>>>((const float4*)x,  xb,  kM * kDModel / 4);
  f32_to_bf16_kernel<<<256,  256, 0, stream>>>((const float4*)Wq, wqb, kDModel * kDModel / 4);
  f32_to_bf16_kernel<<<256,  256, 0, stream>>>((const float4*)Wk, wkb, kDModel * kDModel / 4);
  f32_to_bf16_kernel<<<256,  256, 0, stream>>>((const float4*)Wv, wvb, kDModel * kDModel / 4);
  f32_to_bf16_kernel<<<256,  256, 0, stream>>>((const float4*)Wo, wob, kDModel * kDModel / 4);

  // 2) q/k/v projections (WMMA GEMMs, bf16 out)
  dim3 gGemm(kDModel / 128, kM / 128);  // (8,128)
  gemm_wmma_bf16<true><<<gGemm, 256, 0, stream>>>(xb, wqb, qb, kM, kDModel, kDModel);
  gemm_wmma_bf16<true><<<gGemm, 256, 0, stream>>>(xb, wkb, kb, kM, kDModel, kDModel);
  gemm_wmma_bf16<true><<<gGemm, 256, 0, stream>>>(xb, wvb, vb, kM, kDModel, kDModel);

  // 3) kv = K^T V per (b,h), split over S then reduced (+scale, transposed)
  kv_partial_kernel<<<dim3(kBH, kSplit), 128, 0, stream>>>(kb, vb, kvpart);
  kv_reduce_kernel<<<kBH, 256, 0, stream>>>(kvpart, kvT);

  // 4) attn = q @ kv (scale folded into kv)
  attn_wmma_kernel<<<dim3(kSeq / 128, kBH), 256, 0, stream>>>(qb, kvT, attnb);

  // 5) out = attn @ Wo^T (fp32 out)
  gemm_wmma_bf16<false><<<gGemm, 256, 0, stream>>>(attnb, wob, out, kM, kDModel, kDModel);
}
